// DynamicKMOELayer_57964878627030
// MI455X (gfx1250) — matrix operand order repaired and checked by reading, not compiled, not run on hardware
//
#include <hip/hip_runtime.h>

#define T_TOK   2048
#define DMODEL  1024
#define DFF     4096
#define NEXP    8
#define FINAL_N (2 * 1024 * 1024)

#define BM 128
#define BN 128
#define BK 64
#define KP (BK / 2)            /* 32 k-pairs per tile                        */
#define AQ_STRIDE (KP + 4)     /* 36 uints per A row (m-major, k-pair packed) */
#define BQ_STRIDE (KP + 4)     /* 36 uints per B row (n-major, k-pair packed);
                                  144 B row stride -> 16-byte aligned rows    */

typedef __attribute__((ext_vector_type(16))) __bf16       v16bf;
typedef __attribute__((ext_vector_type(8)))  float        v8f;
typedef __attribute__((ext_vector_type(8)))  unsigned int v8u;

union Frag { v16bf bf; v8u u; };

__device__ __forceinline__ unsigned int f2bf(float f) {
  union { float f; unsigned int u; } v; v.f = f;
  unsigned int r = v.u + 0x7FFFu + ((v.u >> 16) & 1u); // RNE
  return r >> 16;
}
__device__ __forceinline__ unsigned int pack2bf(float lo, float hi) {
  return f2bf(lo) | (f2bf(hi) << 16);
}

// A fragment (16x32 bf16, ISA 7.12.2): M = lane%16,
// k-pair index = (v<4 ? v : v+4) + (lane<16 ? 0 : 4)  [+ 16 per K=32 step]
__device__ __forceinline__ void load_afrag(Frag& fr, const unsigned int* As,
                                           int mrow, int ksPair, int lane) {
  const int half = (lane < 16) ? 0 : 4;
  const unsigned int* p = As + mrow * AQ_STRIDE + ksPair + half;
#pragma unroll
  for (int v = 0; v < 4; ++v) fr.u[v] = p[v];        // contiguous -> ds_load_b128
#pragma unroll
  for (int v = 4; v < 8; ++v) fr.u[v] = p[v + 4];    // contiguous -> ds_load_b128
}

// B fragment (32x16 bf16, ISA 7.12.4 pattern): N = lane%16,
// k-pair index = (v<4 ? v : v+4) + (lane<16 ? 0 : 4)  [+ 16 per K=32 step]
__device__ __forceinline__ void load_bfrag(Frag& fr, const unsigned int* Bs,
                                           int ncol, int ksPair, int lane) {
  const int half = (lane < 16) ? 0 : 4;
  const unsigned int* p = Bs + ncol * BQ_STRIDE + ksPair + half;
#pragma unroll
  for (int v = 0; v < 4; ++v) fr.u[v] = p[v];        // contiguous -> ds_load_b128
#pragma unroll
  for (int v = 4; v < 8; ++v) fr.u[v] = p[v + 4];    // contiguous -> ds_load_b128
}

// ---------------------------------------------------------------------------
// Kernel 1: dynamic-K (top-p) router. One thread per token.
// ---------------------------------------------------------------------------
__global__ void moe_router(const float* __restrict__ x, const float* __restrict__ Wg,
                           const float* __restrict__ bg, float* __restrict__ wf,
                           float* __restrict__ probs, float* __restrict__ ent,
                           float* __restrict__ cnt_out) {
  int t = blockIdx.x * blockDim.x + threadIdx.x;
  if (t >= T_TOK) return;
  float l[NEXP];
#pragma unroll
  for (int e = 0; e < NEXP; ++e) l[e] = bg[e];
  const float* xr = x + (size_t)t * DMODEL;
  for (int d = 0; d < DMODEL; ++d) {
    float xv = xr[d];
    const float* wr = Wg + (size_t)d * NEXP;
#pragma unroll
    for (int e = 0; e < NEXP; ++e) l[e] = fmaf(xv, wr[e], l[e]);
  }
  float m = l[0];
#pragma unroll
  for (int e = 1; e < NEXP; ++e) m = fmaxf(m, l[e]);
  float p[NEXP], s = 0.f;
#pragma unroll
  for (int e = 0; e < NEXP; ++e) { p[e] = __expf(l[e] - m); s += p[e]; }
  float invs = 1.f / s, entv = 0.f;
#pragma unroll
  for (int e = 0; e < NEXP; ++e) { p[e] *= invs; entv -= p[e] * __logf(p[e] + 1e-6f); }
  // stable descending insertion sort (matches argsort(-probs))
  int idx[NEXP]; float sp[NEXP];
#pragma unroll
  for (int e = 0; e < NEXP; ++e) { sp[e] = p[e]; idx[e] = e; }
  for (int i = 1; i < NEXP; ++i) {
    float v = sp[i]; int id = idx[i]; int j = i - 1;
    while (j >= 0 && sp[j] < v) { sp[j + 1] = sp[j]; idx[j + 1] = idx[j]; --j; }
    sp[j + 1] = v; idx[j + 1] = id;
  }
  float excl = 0.f, asum = 0.f, ap[NEXP]; int cnt = 0;
#pragma unroll
  for (int j = 0; j < NEXP; ++j) {
    bool act = (excl < 0.8f);   // j==0: excl==0 -> always active
    excl += sp[j];
    float a = act ? sp[j] : 0.f;
    ap[j] = a; asum += a; cnt += act ? 1 : 0;
  }
  float rn = 1.f / (asum + 1e-6f);
  float wout[NEXP];
#pragma unroll
  for (int e = 0; e < NEXP; ++e) wout[e] = 0.f;
#pragma unroll
  for (int j = 0; j < NEXP; ++j) wout[idx[j]] = ap[j] * rn;
#pragma unroll
  for (int e = 0; e < NEXP; ++e) { wf[t * NEXP + e] = wout[e]; probs[t * NEXP + e] = p[e]; }
  ent[t] = entv;
  cnt_out[t] = (float)cnt;
}

// ---------------------------------------------------------------------------
// Kernel 2: auxiliary-loss reduction (single block).
// ---------------------------------------------------------------------------
__global__ void moe_loss(const float* __restrict__ wf, const float* __restrict__ probs,
                         const float* __restrict__ ent, float* __restrict__ out) {
  __shared__ float sred[256];
  int tid = threadIdx.x;
  float entsum = 0.f, fi[NEXP], Pi[NEXP];
#pragma unroll
  for (int e = 0; e < NEXP; ++e) { fi[e] = 0.f; Pi[e] = 0.f; }
  for (int t = tid; t < T_TOK; t += 256) {
    entsum += ent[t];
#pragma unroll
    for (int e = 0; e < NEXP; ++e) {
      fi[e] += (wf[t * NEXP + e] > 0.f) ? 1.f : 0.f;
      Pi[e] += probs[t * NEXP + e];
    }
  }
  auto red = [&](float v) {
    sred[tid] = v; __syncthreads();
    for (int s2 = 128; s2 > 0; s2 >>= 1) { if (tid < s2) sred[tid] += sred[tid + s2]; __syncthreads(); }
    float r = sred[0]; __syncthreads(); return r;
  };
  float E = red(entsum);
  float lb = 0.f;
  for (int e = 0; e < NEXP; ++e) { float f = red(fi[e]); float P = red(Pi[e]); lb += f * P; }
  if (tid == 0) {
    out[FINAL_N]     = (float)NEXP * lb / ((float)T_TOK * (float)T_TOK);
    out[FINAL_N + 1] = E / (float)T_TOK;
  }
}

// ---------------------------------------------------------------------------
// Kernel 3: fused SwiGLU expert up-projection, routing weight folded in.
//   gw[e,t,f] = bf16( w[t,e] * silu(x@W1+b1) * (x@W3+b3) )
// ---------------------------------------------------------------------------
__launch_bounds__(256)
__global__ void moe_gemm1_swiglu(const float* __restrict__ x, const float* __restrict__ W1,
                                 const float* __restrict__ b1, const float* __restrict__ W3,
                                 const float* __restrict__ b3, const float* __restrict__ wf,
                                 unsigned short* __restrict__ gw) {
  __shared__ unsigned int Asu[BM * AQ_STRIDE];   // [m][kp]
  __shared__ unsigned int Bs1[BN * BQ_STRIDE];   // [n][kp]  (n-major!)
  __shared__ unsigned int Bs3[BN * BQ_STRIDE];
  __shared__ float sw[BM], sb1[BN], sb3[BN];

  const int tid = threadIdx.x, lane = tid & 31, wave = tid >> 5;
  const int wm = wave & 1, wn = wave >> 1;
  const int e = blockIdx.z, t0 = blockIdx.y * BM, f0 = blockIdx.x * BN;

  v8f acc1[4][2], acc3[4][2];
  const v8f vzero = {0.f, 0.f, 0.f, 0.f, 0.f, 0.f, 0.f, 0.f};
#pragma unroll
  for (int i = 0; i < 4; ++i)
#pragma unroll
    for (int j = 0; j < 2; ++j) { acc1[i][j] = vzero; acc3[i][j] = vzero; }

  for (int k0 = 0; k0 < DMODEL; k0 += BK) {
    __syncthreads();
    for (int i = tid; i < BM * KP; i += 256) {        // x tile, fp32->bf16 k-pair packed
      int row = i / KP, cp = i % KP;
      const float* xp = x + (size_t)(t0 + row) * DMODEL + k0 + 2 * cp;
      Asu[row * AQ_STRIDE + cp] = pack2bf(xp[0], xp[1]);
    }
    for (int i = tid; i < KP * BN; i += 256) {        // W1/W3 tiles -> n-major LDS
      int kp = i / BN, n = i % BN;                    // global reads coalesced along n
      size_t base = ((size_t)e * DMODEL + (k0 + 2 * kp)) * DFF + f0 + n;
      Bs1[n * BQ_STRIDE + kp] = pack2bf(W1[base], W1[base + DFF]);
      Bs3[n * BQ_STRIDE + kp] = pack2bf(W3[base], W3[base + DFF]);
    }
    __syncthreads();
#pragma unroll
    for (int ks = 0; ks < 2; ++ks) {                  // two K=32 WMMA steps
      const int ksPair = ks * 16;
      Frag a[4], bA[2], bB[2];
#pragma unroll
      for (int im = 0; im < 4; ++im)
        load_afrag(a[im], Asu, wm * 64 + im * 16 + (lane & 15), ksPair, lane);
#pragma unroll
      for (int in = 0; in < 2; ++in) {
        int ncol = wn * 32 + in * 16 + (lane & 15);
        load_bfrag(bA[in], Bs1, ncol, ksPair, lane);
        load_bfrag(bB[in], Bs3, ncol, ksPair, lane);
      }
#pragma unroll
      for (int im = 0; im < 4; ++im)
#pragma unroll
        for (int in = 0; in < 2; ++in) {
          acc1[im][in] = __builtin_amdgcn_wmma_f32_16x16x32_bf16(
              false, a[im].bf, false, bA[in].bf, (short)0, acc1[im][in], false, false);
          acc3[im][in] = __builtin_amdgcn_wmma_f32_16x16x32_bf16(
              false, a[im].bf, false, bB[in].bf, (short)0, acc3[im][in], false, false);
        }
    }
  }
  __syncthreads();
  if (tid < BM) sw[tid] = wf[(t0 + tid) * NEXP + e];
  if (tid < BN) {
    sb1[tid] = b1[(size_t)e * DFF + f0 + tid];
    sb3[tid] = b3[(size_t)e * DFF + f0 + tid];
  }
  __syncthreads();
#pragma unroll
  for (int im = 0; im < 4; ++im)
#pragma unroll
    for (int in = 0; in < 2; ++in) {
      int N = wn * 32 + in * 16 + (lane & 15);
#pragma unroll
      for (int r = 0; r < 8; ++r) {
        int M = wm * 64 + im * 16 + r + ((lane < 16) ? 0 : 8);
        float h = acc1[im][in][r] + sb1[N];
        float u = acc3[im][in][r] + sb3[N];
        float g = (h / (1.0f + __expf(-h))) * u;
        gw[((size_t)e * T_TOK + t0 + M) * DFF + f0 + N] = (unsigned short)f2bf(g * sw[M]);
      }
    }
}

// ---------------------------------------------------------------------------
// Kernel 4: down-projection + combine.  out = sum_{e,f} gw*W2 + sum_e w*b2
// ---------------------------------------------------------------------------
__launch_bounds__(256)
__global__ void moe_gemm2_combine(const unsigned short* __restrict__ gw,
                                  const float* __restrict__ W2, const float* __restrict__ b2,
                                  const float* __restrict__ wf, float* __restrict__ out) {
  __shared__ unsigned int Asu[BM * AQ_STRIDE];   // [m][kp]
  __shared__ unsigned int Bsp[BN * BQ_STRIDE];   // [n][kp]  (n-major!)
  __shared__ float sw2[BM * NEXP];
  __shared__ float sb2[NEXP * BN];

  const int tid = threadIdx.x, lane = tid & 31, wave = tid >> 5;
  const int wm = wave & 1, wn = wave >> 1;
  const int t0 = blockIdx.y * BM, d0 = blockIdx.x * BN;

  v8f acc[4][2];
  const v8f vzero = {0.f, 0.f, 0.f, 0.f, 0.f, 0.f, 0.f, 0.f};
#pragma unroll
  for (int i = 0; i < 4; ++i)
#pragma unroll
    for (int j = 0; j < 2; ++j) acc[i][j] = vzero;

  for (int e = 0; e < NEXP; ++e) {
    for (int k0 = 0; k0 < DFF; k0 += BK) {
      __syncthreads();
      for (int i = tid; i < BM * KP; i += 256) {    // gw already bf16: raw uint pairs
        int row = i / KP, cp = i % KP;
        Asu[row * AQ_STRIDE + cp] = *(const unsigned int*)(const void*)(
            gw + ((size_t)e * T_TOK + t0 + row) * DFF + k0 + 2 * cp);
      }
      for (int i = tid; i < KP * BN; i += 256) {    // W2 fp32 -> bf16, n-major LDS
        int kp = i / BN, n = i % BN;
        const float* p = W2 + ((size_t)e * DFF + (k0 + 2 * kp)) * DMODEL + d0 + n;
        Bsp[n * BQ_STRIDE + kp] = pack2bf(p[0], p[DMODEL]);
      }
      __syncthreads();
#pragma unroll
      for (int ks = 0; ks < 2; ++ks) {
        const int ksPair = ks * 16;
        Frag a[4], b[2];
#pragma unroll
        for (int im = 0; im < 4; ++im)
          load_afrag(a[im], Asu, wm * 64 + im * 16 + (lane & 15), ksPair, lane);
#pragma unroll
        for (int in = 0; in < 2; ++in)
          load_bfrag(b[in], Bsp, wn * 32 + in * 16 + (lane & 15), ksPair, lane);
#pragma unroll
        for (int im = 0; im < 4; ++im)
#pragma unroll
          for (int in = 0; in < 2; ++in)
            acc[im][in] = __builtin_amdgcn_wmma_f32_16x16x32_bf16(
                false, a[im].bf, false, b[in].bf, (short)0, acc[im][in], false, false);
      }
    }
  }
  __syncthreads();
  for (int i = tid; i < BM * NEXP; i += 256)
    sw2[i] = wf[(size_t)(t0 + i / NEXP) * NEXP + (i % NEXP)];
  for (int i = tid; i < NEXP * BN; i += 256)
    sb2[i] = b2[(size_t)(i / BN) * DMODEL + d0 + (i % BN)];
  __syncthreads();
#pragma unroll
  for (int im = 0; im < 4; ++im)
#pragma unroll
    for (int in = 0; in < 2; ++in) {
      int N = wn * 32 + in * 16 + (lane & 15);
#pragma unroll
      for (int r = 0; r < 8; ++r) {
        int M = wm * 64 + im * 16 + r + ((lane < 16) ? 0 : 8);
        float bias = 0.f;
#pragma unroll
        for (int e = 0; e < NEXP; ++e) bias = fmaf(sw2[M * NEXP + e], sb2[e * BN + N], bias);
        out[(size_t)(t0 + M) * DMODEL + d0 + N] = acc[im][in][r] + bias;
      }
    }
}

// ---------------------------------------------------------------------------
extern "C" void kernel_launch(void* const* d_in, const int* in_sizes, int n_in,
                              void* d_out, int out_size, void* d_ws, size_t ws_size,
                              hipStream_t stream) {
  const float* x  = (const float*)d_in[0];
  const float* Wg = (const float*)d_in[1];
  const float* bg = (const float*)d_in[2];
  const float* W1 = (const float*)d_in[3];
  const float* b1 = (const float*)d_in[4];
  const float* W3 = (const float*)d_in[5];
  const float* b3 = (const float*)d_in[6];
  const float* W2 = (const float*)d_in[7];
  const float* b2 = (const float*)d_in[8];
  float* out = (float*)d_out;

  char* ws = (char*)d_ws;
  float* wf    = (float*)(ws);               // [2048][8] routing weights
  float* probs = (float*)(ws + (64 << 10));  // [2048][8]
  float* ent   = (float*)(ws + (128 << 10)); // [2048]
  unsigned short* gw = (unsigned short*)(ws + (1 << 20)); // bf16 [8][2048][4096] = 128 MB

  moe_router<<<T_TOK / 256, 256, 0, stream>>>(x, Wg, bg, wf, probs, ent, out + FINAL_N + 2);
  moe_loss<<<1, 256, 0, stream>>>(wf, probs, ent, out);
  moe_gemm1_swiglu<<<dim3(DFF / BN, T_TOK / BM, NEXP), 256, 0, stream>>>(x, W1, b1, W3, b3, wf, gw);
  moe_gemm2_combine<<<dim3(DMODEL / BN, T_TOK / BM), 256, 0, stream>>>(gw, W2, b2, wf, out);
}
